// DialogueRNNCell_1443109012098
// MI455X (gfx1250) — matrix operand order repaired and checked by reading, not compiled, not run on hardware
//
#include <hip/hip_runtime.h>

// ---------------------------------------------------------------------------
// DialogueRNN cell for MI455X (gfx1250): bf16 WMMA GEMMs + fused flash-style
// attention. All matrix math uses v_wmma_f32_16x16x32_bf16 (f32 accumulate).
// GEMM: block tile 128x128, 8 wave32s in 4(M)x2(N) grid, wave tile 32x64
// -> 8 WMMAs per K-step per wave with register-level A/B fragment reuse.
// ---------------------------------------------------------------------------

typedef __attribute__((ext_vector_type(16))) __bf16 v16bf;
typedef __attribute__((ext_vector_type(8)))  float  v8f;

#define DEVFN static __device__ __forceinline__

DEVFN unsigned short f2bf(float f) {
  unsigned int u = __builtin_bit_cast(unsigned int, f);
  unsigned int r = u + 0x7FFFu + ((u >> 16) & 1u);   // round-to-nearest-even
  return (unsigned short)(r >> 16);
}

DEVFN float sigmoidf_(float x) { return 1.f / (1.f + __expf(-x)); }

// ---------------------------------------------------------------------------
// Small utility kernels
// ---------------------------------------------------------------------------

__global__ void k_f32_to_bf16(const float* __restrict__ src,
                              unsigned short* __restrict__ dst, int n) {
  int i = blockIdx.x * blockDim.x + threadIdx.x;
  if (i < n) dst[i] = f2bf(src[i]);
}

__global__ void k_argmax8(const float* __restrict__ spk, int* __restrict__ idx, int Bn) {
  int b = blockIdx.x * blockDim.x + threadIdx.x;
  if (b >= Bn) return;
  const float* row = spk + (size_t)b * 8;
  int best = 0; float bv = row[0];
  #pragma unroll
  for (int j = 1; j < 8; ++j) { float v = row[j]; if (v > bv) { bv = v; best = j; } }
  idx[b] = best;
}

// write bf16(feature) into first half of a [B, 2H] concat buffer
__global__ void k_pack_feat(const float* __restrict__ feat,
                            unsigned short* __restrict__ dst) {
  int i = blockIdx.x * blockDim.x + threadIdx.x;   // B*512 threads
  int b = i >> 9, g = i & 511;
  dst[(size_t)b * 1024 + g] = f2bf(feat[i]);
}

// Xg second half: sel_last = last_speaker_state[b, spk_idx[b], :]
__global__ void k_gather_sel_last(const float* __restrict__ lss,
                                  const int* __restrict__ idx,
                                  unsigned short* __restrict__ Xg) {
  int i = blockIdx.x * blockDim.x + threadIdx.x;   // B*512
  int b = i >> 9, g = i & 511;
  float v = lss[((size_t)b * 8 + idx[b]) * 512 + g];
  Xg[(size_t)b * 1024 + 512 + g] = f2bf(v);
}

// ---------------------------------------------------------------------------
// bf16 NT GEMM: C[M,N] = A[M,K] * B[N,K]^T (+ bias[N])
// Requires M%128==0, N%128==0, K%32==0.
// ---------------------------------------------------------------------------

struct Chunk32 { uint4 lo, hi; };   // 32 bytes = 16 bf16

__global__ __launch_bounds__(256)
void k_gemm_bf16_nt(const unsigned short* __restrict__ A,
                    const unsigned short* __restrict__ Bm,
                    const float* __restrict__ bias,
                    float* __restrict__ C,
                    int M, int N, int K) {
  __shared__ unsigned short As[128][40];   // +8 pad, rows stay 16B aligned
  __shared__ unsigned short Bs[128][40];

  const int tid  = threadIdx.x;
  const int wave = tid >> 5;
  const int lane = tid & 31;
  const int mw   = wave >> 1;       // 0..3 : wave row within block tile
  const int nw   = wave & 1;        // 0..1 : wave col within block tile
  const int m0   = blockIdx.y * 128;
  const int n0   = blockIdx.x * 128;
  const int half = lane >> 4;       // K-half selector per ISA 16-bit layouts
  const int r16  = lane & 15;

  v8f acc[2][4];
  #pragma unroll
  for (int s = 0; s < 2; ++s)
    #pragma unroll
    for (int t = 0; t < 4; ++t)
      #pragma unroll
      for (int j = 0; j < 8; ++j) acc[s][t][j] = 0.f;

  for (int kt = 0; kt < K; kt += 32) {
    // stage A tile: 128 rows x 32 cols, 16B chunks (2 per thread)
    #pragma unroll
    for (int it = 0; it < 2; ++it) {
      int c  = tid + it * 256;
      int r  = c >> 2, cc = c & 3;
      *(uint4*)&As[r][cc * 8] =
          *(const uint4*)(A + (size_t)(m0 + r) * K + kt + cc * 8);
    }
    // stage B tile: 128 rows x 32 cols (2 per thread)
    #pragma unroll
    for (int it = 0; it < 2; ++it) {
      int c  = tid + it * 256;
      int r  = c >> 2, cc = c & 3;
      *(uint4*)&Bs[r][cc * 8] =
          *(const uint4*)(Bm + (size_t)(n0 + r) * K + kt + cc * 8);
    }
    __syncthreads();

    // A fragments (16x32 bf16 each): lanes 0-15 hold K 0..7,16..23;
    // lanes 16-31 hold K 8..15,24..31.
    v16bf a[2];
    #pragma unroll
    for (int s = 0; s < 2; ++s) {
      Chunk32 afr;
      afr.lo = *(const uint4*)&As[mw * 32 + s * 16 + r16][half * 8];
      afr.hi = *(const uint4*)&As[mw * 32 + s * 16 + r16][16 + half * 8];
      a[s] = __builtin_bit_cast(v16bf, afr);
    }

    #pragma unroll
    for (int t = 0; t < 4; ++t) {
      // B fragment (32x16 bf16): col n = r16, same K order as A:
      // lanes 0-15 -> K 0..7,16..23; lanes 16-31 -> K 8..15,24..31.
      Chunk32 bfr;
      bfr.lo = *(const uint4*)&Bs[nw * 64 + t * 16 + r16][half * 8];
      bfr.hi = *(const uint4*)&Bs[nw * 64 + t * 16 + r16][16 + half * 8];
      v16bf bb = __builtin_bit_cast(v16bf, bfr);
      #pragma unroll
      for (int s = 0; s < 2; ++s)
        acc[s][t] = __builtin_amdgcn_wmma_f32_16x16x32_bf16(
            false, a[s], false, bb, (short)0, acc[s][t], false, false);
    }
    __syncthreads();
  }

  // C/D layout: VGPR j -> M = j + 8*half ; lane r16 -> N
  #pragma unroll
  for (int t = 0; t < 4; ++t) {
    int col = n0 + nw * 64 + t * 16 + r16;
    float bv = bias ? bias[col] : 0.f;
    #pragma unroll
    for (int s = 0; s < 2; ++s) {
      #pragma unroll
      for (int j = 0; j < 8; ++j) {
        int row = m0 + mw * 32 + s * 16 + half * 8 + j;
        C[(size_t)row * N + col] = acc[s][t][j] + bv;
      }
    }
  }
}

// ---------------------------------------------------------------------------
// Fused attention: one block per batch row b.
// scores[t] = dot(q[b], lgs[t,b,:]); online softmax; context accumulated in
// registers; bf16 context written straight into second half of Xp concat.
// Reads last_global_state exactly once. Per-timestep reduction: wave32
// shfl_xor butterfly + 8 wave partials in LDS (2 barriers / timestep).
// ---------------------------------------------------------------------------

__global__ __launch_bounds__(256)
void k_attention(const float* __restrict__ q,
                 const float* __restrict__ lgs,   // [T,B,G]
                 unsigned short* __restrict__ Xp,
                 int T_, int Bn, int G_) {
  __shared__ float qs[512];
  __shared__ float red[8];
  const int b    = blockIdx.x;
  const int tid  = threadIdx.x;
  const int wv   = tid >> 5;
  const int lane = tid & 31;

  qs[tid]       = q[(size_t)b * G_ + tid];
  qs[tid + 256] = q[(size_t)b * G_ + tid + 256];
  __syncthreads();

  float m = -INFINITY, l = 0.f, acc0 = 0.f, acc1 = 0.f;
  for (int t = 0; t < T_; ++t) {
    const float* v = lgs + ((size_t)t * Bn + b) * G_;
    float v0 = v[tid], v1 = v[tid + 256];
    float partial = qs[tid] * v0 + qs[tid + 256] * v1;
    // wave32 butterfly reduction (no barriers)
    #pragma unroll
    for (int s = 16; s > 0; s >>= 1) partial += __shfl_xor(partial, s, 32);
    if (lane == 0) red[wv] = partial;
    __syncthreads();
    float sc = red[0] + red[1] + red[2] + red[3] +
               red[4] + red[5] + red[6] + red[7];
    __syncthreads();
    float nm = fmaxf(m, sc);
    float scale = __expf(m - nm);
    float p = __expf(sc - nm);
    l = l * scale + p;
    acc0 = acc0 * scale + p * v0;
    acc1 = acc1 * scale + p * v1;
    m = nm;
  }
  float inv = 1.f / l;
  Xp[(size_t)b * 1024 + 512 + tid]       = f2bf(acc0 * inv);
  Xp[(size_t)b * 1024 + 512 + tid + 256] = f2bf(acc1 * inv);
}

// ---------------------------------------------------------------------------
// GRU pointwise kernels
// ---------------------------------------------------------------------------

// gi,gh both [Bn,3H]; hprev [Bn,H]; out [Bn,H]  (used for global + emotion)
__global__ void k_gru_rows(const float* __restrict__ gi, const float* __restrict__ gh,
                           const float* __restrict__ hprev, float* __restrict__ out,
                           int Bn, int H) {
  int i = blockIdx.x * blockDim.x + threadIdx.x;
  if (i >= Bn * H) return;
  int b = i / H, g = i - b * H;
  const float* gib = gi + (size_t)b * 3 * H;
  const float* ghb = gh + (size_t)b * 3 * H;
  float r = sigmoidf_(gib[g] + ghb[g]);
  float z = sigmoidf_(gib[H + g] + ghb[H + g]);
  float n = tanhf(gib[2 * H + g] + r * ghb[2 * H + g]);
  out[i] = (1.f - z) * n + z * hprev[i];
}

// speaker GRU: gi shared across NP (gi[b]), gh per (b,np).
// emits sel_spk (bf16) for the listener concat and the emotion GEMM.
__global__ void k_gru_speaker(const float* __restrict__ gi,  // [B,3H]
                              const float* __restrict__ gh,  // [B*8,3H]
                              const float* __restrict__ lss, // [B*8,H]
                              const int* __restrict__ idx,
                              float* __restrict__ ss_tmp,
                              unsigned short* __restrict__ selspk_b,
                              unsigned short* __restrict__ Xl,
                              int H) {
  int i = blockIdx.x * blockDim.x + threadIdx.x;   // 4096*H
  if (i >= 4096 * H) return;
  int row = i / H, g = i - row * H;
  int b = row >> 3, np = row & 7;
  const float* gib = gi + (size_t)b * 3 * H;
  const float* ghb = gh + (size_t)row * 3 * H;
  float r = sigmoidf_(gib[g] + ghb[g]);
  float z = sigmoidf_(gib[H + g] + ghb[H + g]);
  float n = tanhf(gib[2 * H + g] + r * ghb[2 * H + g]);
  float v = (1.f - z) * n + z * lss[i];
  ss_tmp[i] = v;
  if (np == idx[b]) {
    unsigned short bv = f2bf(v);
    selspk_b[(size_t)b * H + g] = bv;
    Xl[(size_t)b * 2 * H + H + g] = bv;
  }
}

// listener GRU + one-hot mix, writes final speaker_state to d_out
__global__ void k_gru_listener_mix(const float* __restrict__ gi,  // [B,3H]
                                   const float* __restrict__ gh,  // [B*8,3H]
                                   const float* __restrict__ lss,
                                   const int* __restrict__ idx,
                                   const float* __restrict__ ss_tmp,
                                   float* __restrict__ out_ss, int H) {
  int i = blockIdx.x * blockDim.x + threadIdx.x;
  if (i >= 4096 * H) return;
  int row = i / H, g = i - row * H;
  int b = row >> 3, np = row & 7;
  const float* gib = gi + (size_t)b * 3 * H;
  const float* ghb = gh + (size_t)row * 3 * H;
  float r = sigmoidf_(gib[g] + ghb[g]);
  float z = sigmoidf_(gib[H + g] + ghb[H + g]);
  float n = tanhf(gib[2 * H + g] + r * ghb[2 * H + g]);
  float lv = (1.f - z) * n + z * lss[i];
  out_ss[i] = (np == idx[b]) ? ss_tmp[i] : lv;
}

// ---------------------------------------------------------------------------
// Host-side orchestration
// ---------------------------------------------------------------------------

static inline int nblk(int n) { return (n + 255) / 256; }

extern "C" void kernel_launch(void* const* d_in, const int* in_sizes, int n_in,
                              void* d_out, int out_size, void* d_ws, size_t ws_size,
                              hipStream_t stream) {
  (void)in_sizes; (void)n_in; (void)out_size; (void)ws_size;

  const int Bc = 512, Tc = 128, H = 512, H3 = 1536;

  const float* feature = (const float*)d_in[0];
  const float* speaker = (const float*)d_in[1];
  const float* lgs     = (const float*)d_in[2];   // [T,B,G]
  const float* lss     = (const float*)d_in[3];   // [B,8,P]
  const float* le      = (const float*)d_in[4];   // [B,E]
  const float* gW_ih = (const float*)d_in[5];
  const float* gW_hh = (const float*)d_in[6];
  const float* gb_ih = (const float*)d_in[7];
  const float* gb_hh = (const float*)d_in[8];
  const float* pW_ih = (const float*)d_in[9];
  const float* pW_hh = (const float*)d_in[10];
  const float* pb_ih = (const float*)d_in[11];
  const float* pb_hh = (const float*)d_in[12];
  const float* lW_ih = (const float*)d_in[13];
  const float* lW_hh = (const float*)d_in[14];
  const float* lb_ih = (const float*)d_in[15];
  const float* lb_hh = (const float*)d_in[16];
  const float* eW_ih = (const float*)d_in[17];
  const float* eW_hh = (const float*)d_in[18];
  const float* eb_ih = (const float*)d_in[19];
  const float* eb_hh = (const float*)d_in[20];
  const float* attn_W = (const float*)d_in[21];

  float* out_global  = (float*)d_out;                       // [512,512]
  float* out_speaker = out_global + 512 * 512;              // [512,8,512]
  float* out_emotion = out_speaker + 512 * 8 * 512;         // [512,512]

  // ---- workspace bump allocator (256B aligned) ----
  char* w = (char*)d_ws;
  size_t off = 0;
  auto take = [&](size_t bytes) -> void* {
    off = (off + 255) & ~(size_t)255;
    void* p = w + off;
    off += bytes;
    return p;
  };
  typedef unsigned short u16;

  int* spk_idx = (int*)take(512 * sizeof(int));
  u16* gWih_b  = (u16*)take((size_t)H3 * 1024 * 2);
  u16* gWhh_b  = (u16*)take((size_t)H3 * 512 * 2);
  u16* pWih_b  = (u16*)take((size_t)H3 * 1024 * 2);
  u16* pWhh_b  = (u16*)take((size_t)H3 * 512 * 2);
  u16* lWih_b  = (u16*)take((size_t)H3 * 1024 * 2);
  u16* lWhh_b  = (u16*)take((size_t)H3 * 512 * 2);
  u16* eWih_b  = (u16*)take((size_t)H3 * 512 * 2);
  u16* eWhh_b  = (u16*)take((size_t)H3 * 512 * 2);
  u16* attnW_b = (u16*)take((size_t)512 * 512 * 2);
  u16* feat_b  = (u16*)take((size_t)512 * 512 * 2);
  u16* Hg_b    = (u16*)take((size_t)512 * 512 * 2);
  u16* lss_b   = (u16*)take((size_t)4096 * 512 * 2);
  u16* le_b    = (u16*)take((size_t)512 * 512 * 2);
  u16* selspk_b= (u16*)take((size_t)512 * 512 * 2);
  u16* Xg      = (u16*)take((size_t)512 * 1024 * 2);
  u16* Xp      = (u16*)take((size_t)512 * 1024 * 2);
  u16* Xl      = (u16*)take((size_t)512 * 1024 * 2);
  float* qbuf  = (float*)take((size_t)512 * 512 * 4);
  float* ss_tmp= (float*)take((size_t)4096 * 512 * 4);
  float* giA   = (float*)take((size_t)512 * H3 * 4);        // reused 4x
  float* ghA   = (float*)take((size_t)4096 * H3 * 4);       // reused 4x

  // ---- 1. spk_idx ----
  k_argmax8<<<nblk(Bc), 256, 0, stream>>>(speaker, spk_idx, Bc);

  // ---- 2. bf16 conversions (weights + activations) ----
  k_f32_to_bf16<<<nblk(H3 * 1024), 256, 0, stream>>>(gW_ih, gWih_b, H3 * 1024);
  k_f32_to_bf16<<<nblk(H3 * 512),  256, 0, stream>>>(gW_hh, gWhh_b, H3 * 512);
  k_f32_to_bf16<<<nblk(H3 * 1024), 256, 0, stream>>>(pW_ih, pWih_b, H3 * 1024);
  k_f32_to_bf16<<<nblk(H3 * 512),  256, 0, stream>>>(pW_hh, pWhh_b, H3 * 512);
  k_f32_to_bf16<<<nblk(H3 * 1024), 256, 0, stream>>>(lW_ih, lWih_b, H3 * 1024);
  k_f32_to_bf16<<<nblk(H3 * 512),  256, 0, stream>>>(lW_hh, lWhh_b, H3 * 512);
  k_f32_to_bf16<<<nblk(H3 * 512),  256, 0, stream>>>(eW_ih, eWih_b, H3 * 512);
  k_f32_to_bf16<<<nblk(H3 * 512),  256, 0, stream>>>(eW_hh, eWhh_b, H3 * 512);
  k_f32_to_bf16<<<nblk(512 * 512), 256, 0, stream>>>(attn_W, attnW_b, 512 * 512);
  k_f32_to_bf16<<<nblk(512 * 512), 256, 0, stream>>>(feature, feat_b, 512 * 512);
  k_f32_to_bf16<<<nblk(512 * 512), 256, 0, stream>>>(
      lgs + (size_t)(Tc - 1) * Bc * H, Hg_b, 512 * 512);     // h_{t-1} for global GRU
  k_f32_to_bf16<<<nblk(4096 * 512), 256, 0, stream>>>(lss, lss_b, 4096 * 512);
  k_f32_to_bf16<<<nblk(512 * 512),  256, 0, stream>>>(le, le_b, 512 * 512);

  // ---- 3. concat buffers (first halves = feature) ----
  k_pack_feat<<<nblk(512 * 512), 256, 0, stream>>>(feature, Xg);
  k_pack_feat<<<nblk(512 * 512), 256, 0, stream>>>(feature, Xp);
  k_pack_feat<<<nblk(512 * 512), 256, 0, stream>>>(feature, Xl);
  k_gather_sel_last<<<nblk(512 * 512), 256, 0, stream>>>(lss, spk_idx, Xg);

  // ---- 4. global GRU ----
  k_gemm_bf16_nt<<<dim3(H3 / 128, 512 / 128), 256, 0, stream>>>(
      Xg, gWih_b, gb_ih, giA, 512, H3, 1024);
  k_gemm_bf16_nt<<<dim3(H3 / 128, 512 / 128), 256, 0, stream>>>(
      Hg_b, gWhh_b, gb_hh, ghA, 512, H3, 512);
  k_gru_rows<<<nblk(512 * H), 256, 0, stream>>>(
      giA, ghA, lgs + (size_t)(Tc - 1) * Bc * H, out_global, 512, H);

  // ---- 5. attention (context -> Xp second half) ----
  k_gemm_bf16_nt<<<dim3(512 / 128, 512 / 128), 256, 0, stream>>>(
      feat_b, attnW_b, nullptr, qbuf, 512, 512, 512);
  k_attention<<<512, 256, 0, stream>>>(qbuf, lgs, Xp, Tc, Bc, H);

  // ---- 6. speaker GRU (gi broadcast over NP -> batch 512; gh batch 4096) ----
  k_gemm_bf16_nt<<<dim3(H3 / 128, 512 / 128), 256, 0, stream>>>(
      Xp, pWih_b, pb_ih, giA, 512, H3, 1024);
  k_gemm_bf16_nt<<<dim3(H3 / 128, 4096 / 128), 256, 0, stream>>>(
      lss_b, pWhh_b, pb_hh, ghA, 4096, H3, 512);
  k_gru_speaker<<<nblk(4096 * H), 256, 0, stream>>>(
      giA, ghA, lss, spk_idx, ss_tmp, selspk_b, Xl, H);

  // ---- 7. listener GRU + mix ----
  k_gemm_bf16_nt<<<dim3(H3 / 128, 512 / 128), 256, 0, stream>>>(
      Xl, lWih_b, lb_ih, giA, 512, H3, 1024);
  k_gemm_bf16_nt<<<dim3(H3 / 128, 4096 / 128), 256, 0, stream>>>(
      lss_b, lWhh_b, lb_hh, ghA, 4096, H3, 512);
  k_gru_listener_mix<<<nblk(4096 * H), 256, 0, stream>>>(
      giA, ghA, lss, spk_idx, ss_tmp, out_speaker, H);

  // ---- 8. emotion GRU (input = sel_spk) ----
  k_gemm_bf16_nt<<<dim3(H3 / 128, 512 / 128), 256, 0, stream>>>(
      selspk_b, eWih_b, eb_ih, giA, 512, H3, 512);
  k_gemm_bf16_nt<<<dim3(H3 / 128, 512 / 128), 256, 0, stream>>>(
      le_b, eWhh_b, eb_hh, ghA, 512, H3, 512);
  k_gru_rows<<<nblk(512 * H), 256, 0, stream>>>(
      giA, ghA, le, out_emotion, 512, H);
}
